// ChildSumTreeLSTM_82798379532837
// MI455X (gfx1250) — compile-verified, matrix operand order checked by reading, not run on hardware
//
#include <hip/hip_runtime.h>
#include <hip/hip_bf16.h>
#include <math.h>

// Problem constants (from reference)
#define TL_L   64
#define TL_W   8192
#define TL_IN  256
#define TL_MEM 256

typedef __attribute__((ext_vector_type(16))) _Float16 v16h;
typedef __attribute__((ext_vector_type(8)))  float    v8f;

__device__ __forceinline__ v8f wmma_f16(v16h a, v16h b, v8f c) {
    // D = A(16x32 f16) * B(32x16 f16) + C(16x16 f32)
    return __builtin_amdgcn_wmma_f32_16x16x32_f16(
        /*neg_a=*/false, a, /*neg_b=*/false, b,
        /*c_mod=*/(short)0, c, /*reuse_a=*/false, /*reuse_b=*/false);
}

__device__ __forceinline__ float sigmoidf_(float x) {
    return 1.0f / (1.0f + __expf(-x));
}

// Build the 16-bit A fragment (16x32 tile, M rows x K=32) from row-major f32.
// Lane l: row = mbase + (l&15); halves: a[0..7] = K[kbase+off .. +7],
// a[8..15] = K[kbase+16+off .. +7], off = (l>>4)*8.  (ISA 7.12.2 layout)
__device__ __forceinline__ v16h load_a_frag(const float* __restrict__ arow,
                                            int kbase, int aoff) {
    const float4* p0 = (const float4*)(arow + kbase + aoff);
    const float4* p1 = (const float4*)(arow + kbase + 16 + aoff);
    float4 x0 = p0[0], x1 = p0[1], x2 = p1[0], x3 = p1[1];
    v16h a;
    a[0]  = (_Float16)x0.x; a[1]  = (_Float16)x0.y;
    a[2]  = (_Float16)x0.z; a[3]  = (_Float16)x0.w;
    a[4]  = (_Float16)x1.x; a[5]  = (_Float16)x1.y;
    a[6]  = (_Float16)x1.z; a[7]  = (_Float16)x1.w;
    a[8]  = (_Float16)x2.x; a[9]  = (_Float16)x2.y;
    a[10] = (_Float16)x2.z; a[11] = (_Float16)x2.w;
    a[12] = (_Float16)x3.x; a[13] = (_Float16)x3.y;
    a[14] = (_Float16)x3.z; a[15] = (_Float16)x3.w;
    return a;
}

// Pre-swizzled B fragments: per (ntile, ktile) a 32-lane x 16-half block,
// each lane's 16 halves contiguous (32B -> 2x global_load_b128).
__device__ __forceinline__ v16h load_b_frag(const _Float16* __restrict__ Wp,
                                            int ntile, int kk, int lane) {
    return *(const v16h*)(Wp + ((((size_t)ntile * 8 + kk) * 32) + lane) * 16);
}

// ---------------------------------------------------------------------------
// Pack a row-major f32 weight [256, N] into f16 B-fragment layout.
// packed index p = ((ntile*8 + ktile)*32 + lane)*16 + e
// n = ntile*16 + (lane&15); k = ktile*32 + (e<8 ? (lane>>4)*8+e : 16+(lane>>4)*8+e-8)
__global__ void pack_weight_kernel(const float* __restrict__ src,
                                   _Float16* __restrict__ dst, int N) {
    int t = blockIdx.x * blockDim.x + threadIdx.x;
    if (t >= N * 256) return;
    int e     = t & 15;
    int lane  = (t >> 4) & 31;
    int ktile = (t >> 9) & 7;
    int ntile = t >> 12;
    int n  = ntile * 16 + (lane & 15);
    int hi = lane >> 4;
    int k  = ktile * 32 + ((e < 8) ? (hi * 8 + e) : (16 + hi * 8 + (e - 8)));
    dst[t] = (_Float16)src[(size_t)k * N + n];
}

// ---------------------------------------------------------------------------
// h_sum[pidx[j], :] += h_prev[j, :]   (one child row per block, coalesced atomics)
__global__ __launch_bounds__(256)
void scatter_h_kernel(const float* __restrict__ h_prev,
                      const int* __restrict__ pidx,
                      float* __restrict__ h_sum) {
    int j = blockIdx.x;
    int e = threadIdx.x;
    int p = pidx[j];
    atomicAdd(&h_sum[(size_t)p * TL_MEM + e], h_prev[(size_t)j * TL_MEM + e]);
}

// ---------------------------------------------------------------------------
// fxp = x_l @ Wfx + bfx     [8192x256] = [8192x256]@[256x256]
// Block: 256 thr (8 waves), tile M=32 x N=256; wave w -> ntiles {2w, 2w+1},
// two M row-blocks so each B fragment feeds two WMMAs.
__global__ __launch_bounds__(256)
void fxp_kernel(const float* __restrict__ x_l,
                const _Float16* __restrict__ WfxP,
                const float* __restrict__ bfx,
                float* __restrict__ fxp) {
    const int lane = threadIdx.x & 31, w = threadIdx.x >> 5;
    const int lo = lane & 15, hi = lane >> 4;
    const int mbase = blockIdx.x * 32;
    const float* arow0 = x_l + (size_t)(mbase + lo) * TL_IN;
    const float* arow1 = x_l + (size_t)(mbase + 16 + lo) * TL_IN;
    const int aoff = hi * 8;
    const int nt0 = 2 * w, nt1 = 2 * w + 1;
    v8f a00 = {}, a01 = {}, a10 = {}, a11 = {};
#pragma unroll
    for (int kk = 0; kk < 8; ++kk) {
        v16h f0 = load_a_frag(arow0, kk * 32, aoff);
        v16h f1 = load_a_frag(arow1, kk * 32, aoff);
        v16h b0 = load_b_frag(WfxP, nt0, kk, lane);
        v16h b1 = load_b_frag(WfxP, nt1, kk, lane);
        a00 = wmma_f16(f0, b0, a00);
        a10 = wmma_f16(f1, b0, a10);
        a01 = wmma_f16(f0, b1, a01);
        a11 = wmma_f16(f1, b1, a11);
    }
    const int n0 = nt0 * 16 + lo, n1 = nt1 * 16 + lo;
    const float bias0 = bfx[n0], bias1 = bfx[n1];
#pragma unroll
    for (int r = 0; r < 8; ++r) {
        int m0 = mbase + r + hi * 8;
        int m1 = m0 + 16;
        fxp[(size_t)m0 * TL_MEM + n0] = a00[r] + bias0;
        fxp[(size_t)m0 * TL_MEM + n1] = a01[r] + bias1;
        fxp[(size_t)m1 * TL_MEM + n0] = a10[r] + bias0;
        fxp[(size_t)m1 * TL_MEM + n1] = a11[r] + bias1;
    }
}

// ---------------------------------------------------------------------------
// f = sigmoid(h_prev@Wfh + bfh + fxp[pidx]);  fc_sum[pidx] += f * c_prev
__global__ __launch_bounds__(256)
void fgate_kernel(const float* __restrict__ h_prev,
                  const float* __restrict__ c_prev,
                  const _Float16* __restrict__ WfhP,
                  const float* __restrict__ bfh,
                  const float* __restrict__ fxp,
                  const int* __restrict__ pidx,
                  float* __restrict__ fc_sum) {
    const int lane = threadIdx.x & 31, w = threadIdx.x >> 5;
    const int lo = lane & 15, hi = lane >> 4;
    const int mbase = blockIdx.x * 32;
    const float* arow0 = h_prev + (size_t)(mbase + lo) * TL_MEM;
    const float* arow1 = h_prev + (size_t)(mbase + 16 + lo) * TL_MEM;
    const int aoff = hi * 8;
    const int nt0 = 2 * w, nt1 = 2 * w + 1;
    v8f a00 = {}, a01 = {}, a10 = {}, a11 = {};
#pragma unroll
    for (int kk = 0; kk < 8; ++kk) {
        v16h f0 = load_a_frag(arow0, kk * 32, aoff);
        v16h f1 = load_a_frag(arow1, kk * 32, aoff);
        v16h b0 = load_b_frag(WfhP, nt0, kk, lane);
        v16h b1 = load_b_frag(WfhP, nt1, kk, lane);
        a00 = wmma_f16(f0, b0, a00);
        a10 = wmma_f16(f1, b0, a10);
        a01 = wmma_f16(f0, b1, a01);
        a11 = wmma_f16(f1, b1, a11);
    }
    const int n0 = nt0 * 16 + lo, n1 = nt1 * 16 + lo;
    const float bias0 = bfh[n0], bias1 = bfh[n1];
#pragma unroll
    for (int r = 0; r < 8; ++r) {
#pragma unroll
        for (int rb = 0; rb < 2; ++rb) {
            int m = mbase + rb * 16 + r + hi * 8;
            int p = pidx[m];
            float v0 = (rb == 0) ? a00[r] : a10[r];
            float v1 = (rb == 0) ? a01[r] : a11[r];
            float f0 = sigmoidf_(v0 + bias0 + fxp[(size_t)p * TL_MEM + n0]);
            float f1 = sigmoidf_(v1 + bias1 + fxp[(size_t)p * TL_MEM + n1]);
            atomicAdd(&fc_sum[(size_t)p * TL_MEM + n0],
                      f0 * c_prev[(size_t)m * TL_MEM + n0]);
            atomicAdd(&fc_sum[(size_t)p * TL_MEM + n1],
                      f1 * c_prev[(size_t)m * TL_MEM + n1]);
        }
    }
}

// ---------------------------------------------------------------------------
// iou = x_l@Wioux + h_sum@Wiouh + biases; c = sig(i)*tanh(u)+fc_sum;
// h = sig(o)*tanh(c).
// Block: 256 thr, tile M=32. Wave w owns column-triple t = blockIdx.y*8 + w:
// ntiles {t, t+16, t+32}  ==  column n of the i, o and u sections.  Lane
// (r, lo) of the three accumulators then holds the matching i/o/u values, so
// the gate math is pure register arithmetic -- no LDS, no barrier.
__global__ __launch_bounds__(256)
void iou_kernel(const float* __restrict__ x_l,
                const float* __restrict__ h_sum,
                const _Float16* __restrict__ WiouxP,
                const _Float16* __restrict__ WiouhP,
                const float* __restrict__ bioux,
                const float* __restrict__ biouh,
                const float* __restrict__ fc_sum,
                float* __restrict__ out_l,
                float* __restrict__ c_next,
                float* __restrict__ h_next) {
    const int lane = threadIdx.x & 31, w = threadIdx.x >> 5;
    const int lo = lane & 15, hi = lane >> 4;
    const int mbase = blockIdx.x * 32;
    const int t = blockIdx.y * 8 + w;          // 0..15: column triple
    const int aoff = hi * 8;
    const float* ax0 = x_l   + (size_t)(mbase + lo) * TL_IN;
    const float* ax1 = x_l   + (size_t)(mbase + 16 + lo) * TL_IN;
    const float* ah0 = h_sum + (size_t)(mbase + lo) * TL_MEM;
    const float* ah1 = h_sum + (size_t)(mbase + 16 + lo) * TL_MEM;

    v8f ai0 = {}, ao0 = {}, au0 = {}, ai1 = {}, ao1 = {}, au1 = {};
#pragma unroll
    for (int kk = 0; kk < 8; ++kk) {
        v16h f0 = load_a_frag(ax0, kk * 32, aoff);
        v16h f1 = load_a_frag(ax1, kk * 32, aoff);
        v16h bi = load_b_frag(WiouxP, t,      kk, lane);
        v16h bo = load_b_frag(WiouxP, t + 16, kk, lane);
        v16h bu = load_b_frag(WiouxP, t + 32, kk, lane);
        ai0 = wmma_f16(f0, bi, ai0);  ai1 = wmma_f16(f1, bi, ai1);
        ao0 = wmma_f16(f0, bo, ao0);  ao1 = wmma_f16(f1, bo, ao1);
        au0 = wmma_f16(f0, bu, au0);  au1 = wmma_f16(f1, bu, au1);
    }
#pragma unroll
    for (int kk = 0; kk < 8; ++kk) {
        v16h f0 = load_a_frag(ah0, kk * 32, aoff);
        v16h f1 = load_a_frag(ah1, kk * 32, aoff);
        v16h bi = load_b_frag(WiouhP, t,      kk, lane);
        v16h bo = load_b_frag(WiouhP, t + 16, kk, lane);
        v16h bu = load_b_frag(WiouhP, t + 32, kk, lane);
        ai0 = wmma_f16(f0, bi, ai0);  ai1 = wmma_f16(f1, bi, ai1);
        ao0 = wmma_f16(f0, bo, ao0);  ao1 = wmma_f16(f1, bo, ao1);
        au0 = wmma_f16(f0, bu, au0);  au1 = wmma_f16(f1, bu, au1);
    }

    const int n = t * 16 + lo;
    const float bi_ = bioux[n]       + biouh[n];
    const float bo_ = bioux[256 + n] + biouh[256 + n];
    const float bu_ = bioux[512 + n] + biouh[512 + n];
#pragma unroll
    for (int r = 0; r < 8; ++r) {
#pragma unroll
        for (int rb = 0; rb < 2; ++rb) {
            int m = mbase + rb * 16 + r + hi * 8;
            float gi = ((rb == 0) ? ai0[r] : ai1[r]) + bi_;
            float go = ((rb == 0) ? ao0[r] : ao1[r]) + bo_;
            float gu = ((rb == 0) ? au0[r] : au1[r]) + bu_;
            size_t g = (size_t)m * TL_MEM + n;
            float c = sigmoidf_(gi) * tanhf(gu) + fc_sum[g];
            float h = sigmoidf_(go) * tanhf(c);
            out_l[g]  = h;
            c_next[g] = c;
            h_next[g] = h;
        }
    }
}

// ---------------------------------------------------------------------------
extern "C" void kernel_launch(void* const* d_in, const int* in_sizes, int n_in,
                              void* d_out, int out_size, void* d_ws, size_t ws_size,
                              hipStream_t stream) {
    (void)in_sizes; (void)n_in; (void)out_size; (void)ws_size;

    const float* x      = (const float*)d_in[0];   // [L, W, 256]
    const int*   parent = (const int*)  d_in[1];   // [L, W]
    const float* Wioux  = (const float*)d_in[2];   // [256, 768]
    const float* bioux  = (const float*)d_in[3];   // [768]
    const float* Wiouh  = (const float*)d_in[4];   // [256, 768]
    const float* biouh  = (const float*)d_in[5];   // [768]
    const float* Wfx    = (const float*)d_in[6];   // [256, 256]
    const float* bfx    = (const float*)d_in[7];   // [256]
    const float* Wfh    = (const float*)d_in[8];   // [256, 256]
    const float* bfh    = (const float*)d_in[9];   // [256]
    float* out = (float*)d_out;                    // [L, W, 256]

    const size_t SZ  = (size_t)TL_W * TL_MEM;      // floats per state buffer
    const size_t SZB = SZ * sizeof(float);

    char* p = (char*)d_ws;
    auto carve = [&](size_t bytes) -> char* {
        char* r = p;
        p += (bytes + 255) & ~(size_t)255;
        return r;
    };
    float* hb0    = (float*)carve(SZB);
    float* hb1    = (float*)carve(SZB);
    float* cb0    = (float*)carve(SZB);
    float* cb1    = (float*)carve(SZB);
    float* h_sum  = (float*)carve(SZB);
    float* fc_sum = (float*)carve(SZB);
    float* fxp    = (float*)carve(SZB);
    _Float16* WiouxP = (_Float16*)carve((size_t)768 * 256 * 2);
    _Float16* WiouhP = (_Float16*)carve((size_t)768 * 256 * 2);
    _Float16* WfxP   = (_Float16*)carve((size_t)256 * 256 * 2);
    _Float16* WfhP   = (_Float16*)carve((size_t)256 * 256 * 2);

    // Leaf-level child states are zero.
    hipMemsetAsync(hb0, 0, SZB, stream);
    hipMemsetAsync(cb0, 0, SZB, stream);

    // One-time weight swizzle f32 -> f16 B fragments.
    pack_weight_kernel<<<(768 * 256 + 255) / 256, 256, 0, stream>>>(Wioux, WiouxP, 768);
    pack_weight_kernel<<<(768 * 256 + 255) / 256, 256, 0, stream>>>(Wiouh, WiouhP, 768);
    pack_weight_kernel<<<(256 * 256 + 255) / 256, 256, 0, stream>>>(Wfx,   WfxP,   256);
    pack_weight_kernel<<<(256 * 256 + 255) / 256, 256, 0, stream>>>(Wfh,   WfhP,   256);

    for (int l = 0; l < TL_L; ++l) {
        const float* xl   = x + (size_t)l * TL_W * TL_IN;
        const int*   pidx = parent + (size_t)l * TL_W;
        float* hp = (l & 1) ? hb1 : hb0;
        float* cp = (l & 1) ? cb1 : cb0;
        float* hn = (l & 1) ? hb0 : hb1;
        float* cn = (l & 1) ? cb0 : cb1;

        hipMemsetAsync(h_sum, 0, SZB, stream);
        hipMemsetAsync(fc_sum, 0, SZB, stream);

        scatter_h_kernel<<<TL_W, 256, 0, stream>>>(hp, pidx, h_sum);
        fxp_kernel<<<TL_W / 32, 256, 0, stream>>>(xl, WfxP, bfx, fxp);
        fgate_kernel<<<TL_W / 32, 256, 0, stream>>>(hp, cp, WfhP, bfh, fxp, pidx, fc_sum);
        iou_kernel<<<dim3(TL_W / 32, 2), 256, 0, stream>>>(xl, h_sum, WiouxP, WiouhP,
                                                           bioux, biouh, fc_sum,
                                                           out + (size_t)l * TL_W * TL_MEM,
                                                           cn, hn);
    }
}
